// SACB_22960895164697
// MI455X (gfx1250) — matrix operand order repaired and checked by reading, not compiled, not run on hardware
//
#include <hip/hip_runtime.h>

// ---------------------------------------------------------------------------
// Types for CDNA5 WMMA (gfx1250, wave32)
// ---------------------------------------------------------------------------
typedef __bf16 bf16;
typedef __attribute__((ext_vector_type(16))) __bf16 v16bf;
typedef __attribute__((ext_vector_type(8)))  __bf16 v8bf;
typedef __attribute__((ext_vector_type(8)))  float  v8f;

// ---------------------------------------------------------------------------
// Problem constants
// ---------------------------------------------------------------------------
constexpr int DD = 48, HH = 48, WW = 48;
constexpr int CC = 32, OC = 32, K3 = 27;
constexpr size_t NVOX = (size_t)DD * HH * WW;      // 110592
constexpr size_t NCH  = NVOX * CC;                 // 3538944
constexpr float  LP2  = 0.15f * 0.15f;

constexpr int WPP_BYTES = 27 * 1024 * 2;           // 55296  (one weight set, packed bf16)
constexpr int WPF_BYTES = 2 * 27 * 1024 * 2;       // 110592 (two mask weight sets)

// ---------------------------------------------------------------------------
// Workspace layout (bytes, 256-aligned)
// ---------------------------------------------------------------------------
constexpr size_t al256(size_t x) { return (x + 255) & ~(size_t)255; }
constexpr size_t OFF_XBF   = 0;                                   // x, voxel-major bf16 [N][32]
constexpr size_t OFF_WPP   = al256(OFF_XBF   + NVOX * 32 * 2);    // packed proj weights
constexpr size_t OFF_WPF   = al256(OFF_WPP   + WPP_BYTES);        // packed modulated weights (2 masks)
constexpr size_t OFF_XPRAW = al256(OFF_WPF   + WPF_BYTES);        // conv out f32 [32][N]
constexpr size_t OFF_XPF   = al256(OFF_XPRAW + NCH * 4);          // normalized f32 [32][N]
constexpr size_t OFF_XPBF  = al256(OFF_XPF   + NCH * 4);          // normalized bf16 [N][32]
constexpr size_t OFF_AR    = al256(OFF_XPBF  + NVOX * 32 * 2);    // fft ping (re)
constexpr size_t OFF_AI    = al256(OFF_AR    + NCH * 4);
constexpr size_t OFF_BR    = al256(OFF_AI    + NCH * 4);          // fft pong (re)
constexpr size_t OFF_BI    = al256(OFF_BR    + NCH * 4);
constexpr size_t OFF_DFT   = al256(OFF_BI    + NCH * 4);          // 4 * 2304 floats: Fre,Fim,Ire,Iim
constexpr size_t OFF_MU    = al256(OFF_DFT   + 4 * 2304 * 4);     // mu[32], istd[32]
constexpr size_t OFF_HD    = al256(OFF_MU    + 64 * 4);           // high_dom int [N]
constexpr size_t OFF_GACC  = al256(OFF_HD    + NVOX * 4);         // cen accum [2][32] + counts [2]
constexpr size_t OFF_KMOD  = al256(OFF_GACC  + 66 * 4);           // kmod [2][27]
constexpr size_t OFF_BIAS  = al256(OFF_KMOD  + 54 * 4);           // bias [2][32]

// ---------------------------------------------------------------------------
// Async Global -> LDS staging of packed weight fragments (CDNA5 path,
// GLOBAL_LOAD_ASYNC_TO_LDS_B128 tracked by ASYNCcnt). Each thread copies
// 16-byte chunks; wave waits on its own ASYNCcnt, then block barrier.
// ---------------------------------------------------------------------------
__device__ inline void stage_weights_async(const bf16* __restrict__ wp,
                                           char* smem, int totalBytes) {
  unsigned ldsBase = (unsigned)(size_t)smem;            // low 32 bits = LDS offset
  for (int off = threadIdx.x * 16; off < totalBytes; off += 256 * 16) {
    unsigned lds = ldsBase + off;
    unsigned long long g = (unsigned long long)(size_t)((const char*)wp + off);
    asm volatile("global_load_async_to_lds_b128 %0, %1, off"
                 :: "v"(lds), "v"(g) : "memory");
  }
  asm volatile("s_wait_asynccnt 0" ::: "memory");
  __syncthreads();
}

// ---------------------------------------------------------------------------
// A-fragment loader: 16 voxels x 32 channels (bf16) at kernel tap (dz,dy,dx)
// ISA 16-bit A 16x32 layout: lane<16 -> K in {kb..kb+7, kb+16..kb+23}, kb=0;
// lane>=16 -> kb=8. Row M = lane&15.
// ---------------------------------------------------------------------------
__device__ inline v16bf load_A_frag(const bf16* __restrict__ vc,
                                    int z, int y, int x0,
                                    int dz, int dy, int dx, int lane) {
  int m  = lane & 15;
  int kb = (lane >> 4) << 3;                 // 0 or 8
  int zz = z + dz, yy = y + dy, xx = x0 + m + dx;
  v8bf lo = {}, hi = {};
  if ((unsigned)zz < 48u && (unsigned)yy < 48u && (unsigned)xx < 48u) {
    const bf16* p = vc + (size_t)(((zz * 48 + yy) * 48) + xx) * 32;
    lo = *(const v8bf*)(p + kb);             // channels kb..kb+7     (16B aligned)
    hi = *(const v8bf*)(p + kb + 16);        // channels kb+16..kb+23 (16B aligned)
  }
  v16bf a;
#pragma unroll
  for (int i = 0; i < 8; ++i) { a[i] = lo[i]; a[i + 8] = hi[i]; }
  return a;
}

// prefetch the activation row for the next kernel tap (global_prefetch_b8)
__device__ inline void prefetch_A(const bf16* __restrict__ vc,
                                  int z, int y, int x0, int k, int lane) {
  int dz = k / 9 - 1, dy = (k / 3) % 3 - 1, dx = k % 3 - 1;
  int zz = z + dz, yy = y + dy, xx = x0 + (lane & 15) + dx;
  if ((unsigned)zz < 48u && (unsigned)yy < 48u && (unsigned)xx < 48u)
    __builtin_prefetch(vc + (size_t)(((zz * 48 + yy) * 48) + xx) * 32, 0, 1);
}

// ---------------------------------------------------------------------------
// Layout converters / weight packers
// B fragment layout (16-bit K=32 B): lane<16 -> col=lane, K={0..7,16..23} ; lane>=16
// -> col=lane-16, K={8..15,24..31}; element e maps to K = kb+(e&7)+16*(e>>3).
// Fragment flat index: ((k*2 + octile)*32 + lane)*16 + e
// ---------------------------------------------------------------------------
__global__ void k_pack_x(const float* __restrict__ x, bf16* __restrict__ xbf) {
  size_t tid = (size_t)blockIdx.x * blockDim.x + threadIdx.x;
  if (tid >= NCH) return;
  size_t n = tid >> 5; int c = (int)(tid & 31);
  xbf[tid] = (bf16)x[(size_t)c * NVOX + n];
}

__global__ void k_pack_wproj(const float* __restrict__ w, bf16* __restrict__ wp) {
  int tid = blockIdx.x * blockDim.x + threadIdx.x;
  if (tid >= 27 * 1024) return;
  int e = tid & 15, L = (tid >> 4) & 31, t = (tid >> 9) & 1, k = tid >> 10;
  int oc = t * 16 + (L & 15);
  int ci = ((L < 16) ? 0 : 8) + (e & 7) + ((e >> 3) << 4);
  wp[tid] = (bf16)w[(size_t)(oc * 32 + ci) * 27 + k];
}

__global__ void k_pack_wbase(const float* __restrict__ wb,
                             const float* __restrict__ kmod,
                             bf16* __restrict__ wp) {
  int tid = blockIdx.x * blockDim.x + threadIdx.x;
  if (tid >= 2 * 27 * 1024) return;
  int e = tid & 15, L = (tid >> 4) & 31, t = (tid >> 9) & 1;
  int k = (tid >> 10) % 27, m = tid / (27 * 1024);
  int oc = t * 16 + (L & 15);
  int ci = ((L < 16) ? 0 : 8) + (e & 7) + ((e >> 3) << 4);
  wp[tid] = (bf16)(kmod[m * 27 + k] * wb[(size_t)(oc * 32 + ci) * 27 + k]);
}

// ---------------------------------------------------------------------------
// Stage A: 3x3x3 conv (32->32) as implicit GEMM on WMMA.
// One wave = 16 consecutive voxels (one x-row segment) x 32 output channels.
// B fragments staged in LDS via async copies, read back with ds_load_b128.
// ---------------------------------------------------------------------------
__global__ __launch_bounds__(256) void k_conv(const bf16* __restrict__ xbf,
                                              const bf16* __restrict__ wp,
                                              const float* __restrict__ bproj,
                                              float* __restrict__ out) {
  extern __shared__ char smem[];
  stage_weights_async(wp, smem, WPP_BYTES);
  const v16bf* wv = (const v16bf*)smem;

  int lane = threadIdx.x & 31, wave = threadIdx.x >> 5;
  int tb = (blockIdx.x * 8 + wave) * 16;
  int z = tb / 2304, y = (tb / 48) % 48, x0 = tb % 48;
  v8f a0 = {}, a1 = {};
  for (int k = 0; k < 27; ++k) {
    int dz = k / 9 - 1, dy = (k / 3) % 3 - 1, dx = k % 3 - 1;
    v16bf a  = load_A_frag(xbf, z, y, x0, dz, dy, dx, lane);
    if (k < 26) prefetch_A(xbf, z, y, x0, k + 1, lane);
    v16bf b0 = wv[(k * 2 + 0) * 32 + lane];
    v16bf b1 = wv[(k * 2 + 1) * 32 + lane];
    a0 = __builtin_amdgcn_wmma_f32_16x16x32_bf16(false, a, false, b0, (short)0, a0, false, false);
    a1 = __builtin_amdgcn_wmma_f32_16x16x32_bf16(false, a, false, b1, (short)0, a1, false, false);
  }
  int col = lane & 15, mb = (lane >> 4) * 8;
#pragma unroll
  for (int r = 0; r < 8; ++r) {
    int v = tb + mb + r;
    out[(size_t)col * NVOX + v]        = a0[r] + bproj[col];
    out[(size_t)(col + 16) * NVOX + v] = a1[r] + bproj[col + 16];
  }
}

// ---------------------------------------------------------------------------
// Instance norm statistics / apply (+PReLU), writing f32 [C][N] and bf16 [N][C]
// ---------------------------------------------------------------------------
__global__ void k_stats(const float* __restrict__ xr, float* __restrict__ mu,
                        float* __restrict__ istd) {
  int c = blockIdx.x, t = threadIdx.x;
  float s = 0.f, s2 = 0.f;
  for (size_t i = t; i < NVOX; i += 256) {
    float v = xr[(size_t)c * NVOX + i];
    s += v; s2 += v * v;
  }
  __shared__ float ss[256], sq[256];
  ss[t] = s; sq[t] = s2; __syncthreads();
  for (int w = 128; w > 0; w >>= 1) {
    if (t < w) { ss[t] += ss[t + w]; sq[t] += sq[t + w]; }
    __syncthreads();
  }
  if (t == 0) {
    float mean = ss[0] / (float)NVOX;
    float var  = sq[0] / (float)NVOX - mean * mean;
    mu[c] = mean; istd[c] = rsqrtf(var + 1e-5f);
  }
}

__global__ void k_apply(const float* __restrict__ xr, const float* __restrict__ mu,
                        const float* __restrict__ istd, const float* __restrict__ ap,
                        float* __restrict__ xpf, bf16* __restrict__ xpbf) {
  size_t tid = (size_t)blockIdx.x * blockDim.x + threadIdx.x;
  if (tid >= NCH) return;
  int c = (int)(tid / NVOX); size_t n = tid % NVOX;
  float v = (xr[tid] - mu[c]) * istd[c];
  v = (v >= 0.f) ? v : ap[0] * v;
  xpf[tid] = v;
  xpbf[n * 32 + c] = (bf16)v;
}

// ---------------------------------------------------------------------------
// FFT via dense 48-point DFT matrices, one axis per pass.
// ---------------------------------------------------------------------------
__global__ void k_dft_init(float* __restrict__ d) {
  int tid = blockIdx.x * blockDim.x + threadIdx.x;
  if (tid >= 2304) return;
  int j = tid / 48, n = tid % 48;
  double ang = -2.0 * 3.14159265358979323846 * (double)(j * n) / 48.0;
  d[tid]            = (float)cos(ang);          // fwd re
  d[2304 + tid]     = (float)sin(ang);          // fwd im
  d[2 * 2304 + tid] = (float)(cos(ang) / 48.0); // inv re (cos even)
  d[3 * 2304 + tid] = (float)(-sin(ang) / 48.0);// inv im (conj, /48)
}

__global__ __launch_bounds__(64) void k_dft_pass(const float* __restrict__ sRe,
                                                 const float* __restrict__ sIm,
                                                 float* __restrict__ dRe,
                                                 float* __restrict__ dIm,
                                                 const float* __restrict__ mre,
                                                 const float* __restrict__ mim,
                                                 int axis, int realIn) {
  __shared__ float lre[48], lim[48];
  int t = threadIdx.x;
  int line = blockIdx.x;
  int c = line / 2304, rem = line % 2304;
  size_t base; int stride;
  if (axis == 0)      { base = (size_t)c * NVOX + (size_t)rem * 48; stride = 1; }
  else if (axis == 1) { base = (size_t)c * NVOX + (size_t)(rem / 48) * 2304 + (rem % 48); stride = 48; }
  else                { base = (size_t)c * NVOX + rem; stride = 2304; }
  if (t < 48) {
    lre[t] = sRe[base + (size_t)t * stride];
    lim[t] = realIn ? 0.f : sIm[base + (size_t)t * stride];
  }
  __syncthreads();
  if (t < 48) {
    float re = 0.f, im = 0.f;
    for (int n = 0; n < 48; ++n) {
      float wr = mre[t * 48 + n], wi = mim[t * 48 + n];
      float xr = lre[n], xi = lim[n];
      re += wr * xr - wi * xi;
      im += wr * xi + wi * xr;
    }
    dRe[base + (size_t)t * stride] = re;
    dIm[base + (size_t)t * stride] = im;
  }
}

// spherical low-pass mask in fftshifted coords; zero high freqs in-place
__global__ void k_freq_mask(float* __restrict__ re, float* __restrict__ im) {
  size_t tid = (size_t)blockIdx.x * blockDim.x + threadIdx.x;
  if (tid >= NCH) return;
  size_t n = tid % NVOX;
  int fz = (int)(n / 2304), fy = (int)((n / 48) % 48), fx = (int)(n % 48);
  float gz = -1.f + (float)((fz + 24) % 48) * (2.f / 47.f);
  float gy = -1.f + (float)((fy + 24) % 48) * (2.f / 47.f);
  float gx = -1.f + (float)((fx + 24) % 48) * (2.f / 47.f);
  if (gz * gz + gy * gy + gx * gx > LP2) { re[tid] = 0.f; im[tid] = 0.f; }
}

// low = mean_c |low_field|, high = mean_c |xp - low_field|  ->  high_dom mask
__global__ void k_fields(const float* __restrict__ lr, const float* __restrict__ li,
                         const float* __restrict__ xpf, int* __restrict__ hd) {
  size_t n = (size_t)blockIdx.x * blockDim.x + threadIdx.x;
  if (n >= NVOX) return;
  float lo = 0.f, hi = 0.f;
  for (int c = 0; c < 32; ++c) {
    size_t i = (size_t)c * NVOX + n;
    float re = lr[i], im = li[i];
    lo += sqrtf(re * re + im * im);
    float hr = xpf[i] - re;
    hi += sqrtf(hr * hr + im * im);
  }
  hd[n] = (hi > lo) ? 1 : 0;
}

// separable 3-tap box sum (zero padded) along one axis
__global__ void k_box(const float* __restrict__ src, float* __restrict__ dst, int axis) {
  size_t tid = (size_t)blockIdx.x * blockDim.x + threadIdx.x;
  if (tid >= NCH) return;
  size_t n = tid % NVOX;
  int z = (int)(n / 2304), y = (int)((n / 48) % 48), x = (int)(n % 48);
  int q = (axis == 0) ? x : (axis == 1) ? y : z;
  size_t s = (axis == 0) ? 1 : (axis == 1) ? 48 : 2304;
  float v = src[tid];
  if (q > 0)  v += src[tid - s];
  if (q < 47) v += src[tid + s];
  dst[tid] = v;
}

__global__ void k_zero(float* __restrict__ g) {
  int t = threadIdx.x;
  if (t < 66) g[t] = 0.f;
}

// centroid accumulation: gacc[m*32+c] = sum_n mask_m(n) * nbsum[c][n]; counts at [64],[65]
__global__ void k_cen_reduce(const float* __restrict__ nb, const int* __restrict__ hd,
                             float* __restrict__ g) {
  __shared__ float sh[66];
  int t = threadIdx.x;
  if (t < 66) sh[t] = 0.f;
  __syncthreads();
  float a0[32], a1[32];
#pragma unroll
  for (int c = 0; c < 32; ++c) { a0[c] = 0.f; a1[c] = 0.f; }
  float c0 = 0.f, c1 = 0.f;
  for (size_t n = (size_t)blockIdx.x * 256 + t; n < NVOX; n += 64 * 256) {
    int m = hd[n];
    if (m) c1 += 1.f; else c0 += 1.f;
#pragma unroll
    for (int c = 0; c < 32; ++c) {
      float v = nb[(size_t)c * NVOX + n];
      a0[c] += m ? 0.f : v;
      a1[c] += m ? v : 0.f;
    }
  }
#pragma unroll
  for (int c = 0; c < 32; ++c) {
    atomicAdd(&sh[c], a0[c]);
    atomicAdd(&sh[32 + c], a1[c]);
  }
  atomicAdd(&sh[64], c0);
  atomicAdd(&sh[65], c1);
  __syncthreads();
  if (t < 66) atomicAdd(&g[t], sh[t]);
}

// tiny MLPs: cen -> kmod[2][27] (sigmoid), bias[2][32]
__global__ __launch_bounds__(128) void k_mlp(const float* __restrict__ g,
    const float* kw1, const float* kb1, const float* kw2, const float* kb2,
    const float* kw3, const float* kb3, const float* bw1, const float* bb1,
    const float* bw2, const float* bb2, const float* bw3, const float* bb3,
    float* __restrict__ kmod, float* __restrict__ biasv) {
  __shared__ float cen[2][32], h1[128], h2[128], g1[64], g2[64];
  int t = threadIdx.x;
  if (t < 64) {
    int m = t >> 5, c = t & 31;
    cen[m][c] = g[t] / (27.f * fmaxf(g[64 + m], 1.f));
  }
  __syncthreads();
  for (int m = 0; m < 2; ++m) {
    float s = kb1[t];
    for (int c = 0; c < 32; ++c) s += cen[m][c] * kw1[c * 128 + t];
    h1[t] = fmaxf(s, 0.f);
    __syncthreads();
    s = kb2[t];
    for (int j = 0; j < 128; ++j) s += h1[j] * kw2[j * 128 + t];
    h2[t] = fmaxf(s, 0.f);
    __syncthreads();
    if (t < 27) {
      s = kb3[t];
      for (int j = 0; j < 128; ++j) s += h2[j] * kw3[j * 27 + t];
      kmod[m * 27 + t] = 1.f / (1.f + expf(-s));
    }
    if (t < 64) {
      s = bb1[t];
      for (int c = 0; c < 32; ++c) s += cen[m][c] * bw1[c * 64 + t];
      g1[t] = fmaxf(s, 0.f);
    }
    __syncthreads();
    if (t < 64) {
      s = bb2[t];
      for (int j = 0; j < 64; ++j) s += g1[j] * bw2[j * 64 + t];
      g2[t] = fmaxf(s, 0.f);
    }
    __syncthreads();
    if (t < 32) {
      s = bb3[t];
      for (int j = 0; j < 64; ++j) s += g2[j] * bw3[j * 32 + t];
      biasv[m * 32 + t] = s;
    }
    __syncthreads();
  }
}

// ---------------------------------------------------------------------------
// Main modulated conv: both mask-weight sets in one K-loop (shared A loads),
// weights staged in LDS via async copies, per-voxel mask select + bias +
// PReLU + residual in a fused epilogue.
// ---------------------------------------------------------------------------
__global__ __launch_bounds__(256) void k_main(const bf16* __restrict__ xpbf,
                                              const bf16* __restrict__ wp,
                                              const int* __restrict__ hd,
                                              const float* __restrict__ biasv,
                                              const float* __restrict__ aop,
                                              const float* __restrict__ xres,
                                              float* __restrict__ out) {
  extern __shared__ char smem[];
  stage_weights_async(wp, smem, WPF_BYTES);
  const v16bf* wv = (const v16bf*)smem;

  int lane = threadIdx.x & 31, wave = threadIdx.x >> 5;
  int tb = (blockIdx.x * 8 + wave) * 16;
  int z = tb / 2304, y = (tb / 48) % 48, x0 = tb % 48;
  v8f l0 = {}, l1 = {}, h0 = {}, h1 = {};
  for (int k = 0; k < 27; ++k) {
    int dz = k / 9 - 1, dy = (k / 3) % 3 - 1, dx = k % 3 - 1;
    v16bf a   = load_A_frag(xpbf, z, y, x0, dz, dy, dx, lane);
    if (k < 26) prefetch_A(xpbf, z, y, x0, k + 1, lane);
    v16bf bL0 = wv[(k * 2 + 0) * 32 + lane];
    v16bf bL1 = wv[(k * 2 + 1) * 32 + lane];
    v16bf bH0 = wv[1728 + (k * 2 + 0) * 32 + lane];
    v16bf bH1 = wv[1728 + (k * 2 + 1) * 32 + lane];
    l0 = __builtin_amdgcn_wmma_f32_16x16x32_bf16(false, a, false, bL0, (short)0, l0, false, false);
    l1 = __builtin_amdgcn_wmma_f32_16x16x32_bf16(false, a, false, bL1, (short)0, l1, false, false);
    h0 = __builtin_amdgcn_wmma_f32_16x16x32_bf16(false, a, false, bH0, (short)0, h0, false, false);
    h1 = __builtin_amdgcn_wmma_f32_16x16x32_bf16(false, a, false, bH1, (short)0, h1, false, false);
  }
  float ao = aop[0];
  int col = lane & 15, mb = (lane >> 4) * 8;
#pragma unroll
  for (int r = 0; r < 8; ++r) {
    int v = tb + mb + r;
    int m = hd[v];
    float r0 = (m ? h0[r] : l0[r]) + biasv[m * 32 + col];
    float r1 = (m ? h1[r] : l1[r]) + biasv[m * 32 + col + 16];
    r0 = (r0 >= 0.f) ? r0 : ao * r0;
    r1 = (r1 >= 0.f) ? r1 : ao * r1;
    out[(size_t)col * NVOX + v]        = r0 + xres[(size_t)col * NVOX + v];
    out[(size_t)(col + 16) * NVOX + v] = r1 + xres[(size_t)(col + 16) * NVOX + v];
  }
}

// ---------------------------------------------------------------------------
// Host launcher
// ---------------------------------------------------------------------------
extern "C" void kernel_launch(void* const* d_in, const int* in_sizes, int n_in,
                              void* d_out, int out_size, void* d_ws, size_t ws_size,
                              hipStream_t stream) {
  const float* x       = (const float*)d_in[0];
  const float* w_proj  = (const float*)d_in[1];
  const float* b_proj  = (const float*)d_in[2];
  const float* a_proj  = (const float*)d_in[3];
  const float* w_base  = (const float*)d_in[4];
  const float* kw1 = (const float*)d_in[5],  *kb1 = (const float*)d_in[6];
  const float* kw2 = (const float*)d_in[7],  *kb2 = (const float*)d_in[8];
  const float* kw3 = (const float*)d_in[9],  *kb3 = (const float*)d_in[10];
  const float* bw1 = (const float*)d_in[11], *bb1 = (const float*)d_in[12];
  const float* bw2 = (const float*)d_in[13], *bb2 = (const float*)d_in[14];
  const float* bw3 = (const float*)d_in[15], *bb3 = (const float*)d_in[16];
  const float* a_out   = (const float*)d_in[17];
  float* out = (float*)d_out;
  char*  ws  = (char*)d_ws;

  bf16*  xbf   = (bf16*)(ws + OFF_XBF);
  bf16*  wpP   = (bf16*)(ws + OFF_WPP);
  bf16*  wpF   = (bf16*)(ws + OFF_WPF);
  float* xpraw = (float*)(ws + OFF_XPRAW);
  float* xpf   = (float*)(ws + OFF_XPF);
  bf16*  xpbf  = (bf16*)(ws + OFF_XPBF);
  float* AR    = (float*)(ws + OFF_AR);
  float* AI    = (float*)(ws + OFF_AI);
  float* BR    = (float*)(ws + OFF_BR);
  float* BI    = (float*)(ws + OFF_BI);
  float* dft   = (float*)(ws + OFF_DFT);
  float* Fre = dft, *Fim = dft + 2304, *Ire = dft + 2 * 2304, *Iim = dft + 3 * 2304;
  float* mu    = (float*)(ws + OFF_MU);
  float* istd  = mu + 32;
  int*   hd    = (int*)(ws + OFF_HD);
  float* gacc  = (float*)(ws + OFF_GACC);
  float* kmod  = (float*)(ws + OFF_KMOD);
  float* biasv = (float*)(ws + OFF_BIAS);

  const int ELEMB = (int)((NCH + 255) / 256);      // 13824 blocks for per-element passes

  // Stage A: proj conv (WMMA implicit GEMM, weights async-staged in LDS)
  k_pack_x    <<<ELEMB, 256, 0, stream>>>(x, xbf);
  k_pack_wproj<<<108,   256, 0, stream>>>(w_proj, wpP);
  k_conv      <<<864,   256, WPP_BYTES, stream>>>(xbf, wpP, b_proj, xpraw);

  // Stage B: instance norm + PReLU
  k_stats<<<32,    256, 0, stream>>>(xpraw, mu, istd);
  k_apply<<<ELEMB, 256, 0, stream>>>(xpraw, mu, istd, a_proj, xpf, xpbf);

  // Stage C: FFT partition (3 fwd DFT passes, spherical mask, 3 inv passes)
  k_dft_init<<<9, 256, 0, stream>>>(dft);
  k_dft_pass<<<73728, 64, 0, stream>>>(xpf, xpf, AR, AI, Fre, Fim, 0, 1);
  k_dft_pass<<<73728, 64, 0, stream>>>(AR, AI, BR, BI, Fre, Fim, 1, 0);
  k_dft_pass<<<73728, 64, 0, stream>>>(BR, BI, AR, AI, Fre, Fim, 2, 0);
  k_freq_mask<<<ELEMB, 256, 0, stream>>>(AR, AI);
  k_dft_pass<<<73728, 64, 0, stream>>>(AR, AI, BR, BI, Ire, Iim, 0, 0);
  k_dft_pass<<<73728, 64, 0, stream>>>(BR, BI, AR, AI, Ire, Iim, 1, 0);
  k_dft_pass<<<73728, 64, 0, stream>>>(AR, AI, BR, BI, Ire, Iim, 2, 0);
  k_fields<<<432, 256, 0, stream>>>(BR, BI, xpf, hd);

  // Stage D: 3x3x3 neighborhood sums (separable) + masked centroids
  k_box<<<ELEMB, 256, 0, stream>>>(xpf, AR, 0);
  k_box<<<ELEMB, 256, 0, stream>>>(AR,  AI, 1);
  k_box<<<ELEMB, 256, 0, stream>>>(AI,  AR, 2);
  k_zero<<<1, 128, 0, stream>>>(gacc);
  k_cen_reduce<<<64, 256, 0, stream>>>(AR, hd, gacc);

  // Stage E: MLPs -> kmod, bias; pack modulated WMMA weights
  k_mlp<<<1, 128, 0, stream>>>(gacc, kw1, kb1, kw2, kb2, kw3, kb3,
                               bw1, bb1, bw2, bb2, bw3, bb3, kmod, biasv);
  k_pack_wbase<<<216, 256, 0, stream>>>(w_base, kmod, wpF);

  // Stage F: modulated conv (WMMA, both partitions, async-staged weights)
  k_main<<<864, 256, WPF_BYTES, stream>>>(xpbf, wpF, hd, biasv, a_out, x, out);
}
